// CausalSelfAttention_10797547782454
// MI455X (gfx1250) — matrix operand order repaired and checked by reading, not compiled, not run on hardware
//
#include <hip/hip_runtime.h>

// ---------------------------------------------------------------------------
// Causal self-attention forward for MI455X (gfx1250), wave32 + WMMA f16 + TDM.
//   B=2, T=2048, C=1024, H=16, d=64
// Pipeline:
//   0) convert     : x, Wq, Wk, Wv, Wo  f32 -> f16 (once)
//   1) qkv_gemm    : Q = x Wq^T, K = x Wk^T, V = x Wv^T (TDM-staged, f16 WMMA)
//                    Q,K stored [B,H,T,64]; V stored transposed [B,H,64,T]
//   2) attn        : flash-style causal softmax(QK^T/8) V  -> f16 [B,T,C]
//   3) out_proj    : out = A Wo^T + bo (TDM-staged, f32 out)
// ---------------------------------------------------------------------------

typedef __attribute__((ext_vector_type(16))) _Float16 v16h;
typedef __attribute__((ext_vector_type(8)))  _Float16 v8h;
typedef __attribute__((ext_vector_type(8)))  float    v8f;
typedef __attribute__((ext_vector_type(4)))  unsigned v4u;
typedef __attribute__((ext_vector_type(8)))  int      v8i;
typedef __attribute__((ext_vector_type(4)))  int      v4i;

#define NTOK 4096   // B*T
#define CDIM 1024
#define TSEQ 2048
#define NHEAD 16
#define HDIM 64
#define KSTAGE 64

__device__ __forceinline__ v16h ld_frag(const _Float16* p0, const _Float16* p1) {
  v8h a = *(const v8h*)p0;
  v8h b = *(const v8h*)p1;
  return __builtin_shufflevector(a, b, 0,1,2,3,4,5,6,7,8,9,10,11,12,13,14,15);
}

__device__ __forceinline__ v8f wmma16(v16h a, v16h b, v8f c) {
  return __builtin_amdgcn_wmma_f32_16x16x32_f16(
      false, a, false, b, (short)0, c, false, false);
}

// --- Tensor Data Mover: 2-D f16 tile (tile_d0 x tile_d1) global -> LDS ------
// D# packing per CDNA5 ISA:
//  g0: [1:0]=count=1, [63:32]=lds_addr, [120:64]=global_addr, [127:126]=type=2
//  g1: [17:16]=data_size=1 (2B), [79:48]=tensor_dim0, [111:80]=tensor_dim1,
//      [127:112]=tile_dim0, [143:128]=tile_dim1, [207:160]=tensor_dim0_stride
// 6-arg builtin form (clang-23 / therock): (v4u, v8i, v4i, v4i, v8i, i32 cpol)
__device__ __forceinline__ void tdm_load_2d(unsigned lds_addr, const _Float16* gptr,
                                            unsigned tensor_d0, unsigned tensor_d1,
                                            unsigned stride0,
                                            unsigned tile_d0, unsigned tile_d1) {
  unsigned long long ga = (unsigned long long)(size_t)gptr;
  v4u g0;
  g0[0] = 1u;                                            // count=1, is_restore=0
  g0[1] = lds_addr;                                      // LDS byte address
  g0[2] = (unsigned)(ga & 0xffffffffu);                  // global addr [31:0]
  g0[3] = (unsigned)((ga >> 32) & 0x01ffffffu) | (2u << 30);  // [56:32] | type=2
  v8i g1;
  g1[0] = (int)(1u << 16);                               // wg_mask=0, data_size=2B
  g1[1] = (int)((tensor_d0 & 0xffffu) << 16);            // tensor_dim0 [15:0]
  g1[2] = (int)((tensor_d0 >> 16) | ((tensor_d1 & 0xffffu) << 16));
  g1[3] = (int)((tensor_d1 >> 16) | (tile_d0 << 16));    // tile_dim0
  g1[4] = (int)(tile_d1 & 0xffffu);                      // tile_dim1, tile_dim2=0
  g1[5] = (int)stride0;                                  // tensor_dim0_stride lo
  g1[6] = 0;
  g1[7] = 0;
  v4i z4 = {0, 0, 0, 0};
  v8i z8 = {0, 0, 0, 0, 0, 0, 0, 0};
  __builtin_amdgcn_tensor_load_to_lds(g0, g1, z4, z4, z8, 0);
}

// ---------------------------------------------------------------------------
// Kernel 0: one-shot f32 -> f16 conversion of x and the four weight matrices.
// grid = 2048 (x) + 4*512 (weights) blocks, 256 thr, 8 elems/thread.
// ---------------------------------------------------------------------------
__global__ __launch_bounds__(256) void convert_f32_f16_kernel(
    const float* __restrict__ x,  const float* __restrict__ Wq,
    const float* __restrict__ Wk, const float* __restrict__ Wv,
    const float* __restrict__ Wo,
    _Float16* __restrict__ xh,  _Float16* __restrict__ Wqh,
    _Float16* __restrict__ Wkh, _Float16* __restrict__ Wvh,
    _Float16* __restrict__ Woh)
{
  const int blk = blockIdx.x;
  const float* src; _Float16* dst; size_t base;
  if (blk < 2048) { src = x; dst = xh; base = (size_t)blk * 2048; }
  else {
    const int w = (blk - 2048) >> 9;
    const int r = (blk - 2048) & 511;
    base = (size_t)r * 2048;
    src = (w == 0) ? Wq : (w == 1) ? Wk : (w == 2) ? Wv : Wo;
    dst = (w == 0) ? Wqh : (w == 1) ? Wkh : (w == 2) ? Wvh : Woh;
  }
  const size_t i = base + (size_t)threadIdx.x * 8;
  float4 a = *(const float4*)(src + i);
  float4 b = *(const float4*)(src + i + 4);
  v8h o;
  o[0] = (_Float16)a.x; o[1] = (_Float16)a.y; o[2] = (_Float16)a.z; o[3] = (_Float16)a.w;
  o[4] = (_Float16)b.x; o[5] = (_Float16)b.y; o[6] = (_Float16)b.z; o[7] = (_Float16)b.w;
  *(v8h*)(dst + i) = o;
}

// ---------------------------------------------------------------------------
// Kernel 1: QKV projections. grid = (N/64, M/128, 3), block = 256 (8 waves).
// Block tile 128x64; K-stage 64, double-buffered LDS filled by TDM (wave 0
// issues descriptors; TENSORcnt pairs retire in order -> wait <=2 covers the
// current stage while the next stage streams in).
// ---------------------------------------------------------------------------
__global__ __launch_bounds__(256) void qkv_gemm_kernel(
    const _Float16* __restrict__ xh,
    const _Float16* __restrict__ Wqh, const _Float16* __restrict__ Wkh,
    const _Float16* __restrict__ Wvh,
    _Float16* __restrict__ Qh, _Float16* __restrict__ Kh,
    _Float16* __restrict__ Vt)
{
  __shared__ __align__(16) _Float16 As[2][128][KSTAGE];
  __shared__ __align__(16) _Float16 Bs[2][64][KSTAGE];

  const int which = blockIdx.z;
  const _Float16* W = (which == 0) ? Wqh : ((which == 1) ? Wkh : Wvh);
  const int n0 = blockIdx.x * 64;
  const int m0 = blockIdx.y * 128;
  const int tid  = threadIdx.x;
  const int wave = tid >> 5;
  const int lane = tid & 31;
  const int wm = wave & 3;
  const int wn = wave >> 2;

  const int arow    = lane & 15;
  const int halfsel = lane >> 4;
  const int akb     = halfsel ? 8 : 0;
  const int bkb     = halfsel ? 8 : 0;

  if (wave == 0) {
    tdm_load_2d((unsigned)(size_t)&As[0][0][0], xh + (size_t)m0 * CDIM,
                CDIM, 128, CDIM, KSTAGE, 128);
    tdm_load_2d((unsigned)(size_t)&Bs[0][0][0], W + (size_t)n0 * CDIM,
                CDIM, 64, CDIM, KSTAGE, 64);
  }

  v8f acc[2][2] = {};

  for (int kt = 0; kt < CDIM / KSTAGE; ++kt) {
    const int buf = kt & 1;
    if (wave == 0) {
      if (kt + 1 < CDIM / KSTAGE) {
        const int k1 = (kt + 1) * KSTAGE;
        tdm_load_2d((unsigned)(size_t)&As[buf ^ 1][0][0],
                    xh + (size_t)m0 * CDIM + k1, CDIM, 128, CDIM, KSTAGE, 128);
        tdm_load_2d((unsigned)(size_t)&Bs[buf ^ 1][0][0],
                    W + (size_t)n0 * CDIM + k1, CDIM, 64, CDIM, KSTAGE, 64);
        __builtin_amdgcn_s_wait_tensorcnt(2);   // current stage's pair done
      } else {
        __builtin_amdgcn_s_wait_tensorcnt(0);
      }
    }
    __syncthreads();

#pragma unroll
    for (int ks = 0; ks < 2; ++ks) {
#pragma unroll
      for (int mi = 0; mi < 2; ++mi) {
        const _Float16* ap = &As[buf][wm * 32 + mi * 16 + arow][ks * 32];
        v16h a = ld_frag(ap + akb, ap + akb + 16);
#pragma unroll
        for (int ni = 0; ni < 2; ++ni) {
          const _Float16* bp = &Bs[buf][wn * 32 + ni * 16 + arow][ks * 32];
          v16h b = ld_frag(bp + bkb, bp + bkb + 16);
          acc[mi][ni] = wmma16(a, b, acc[mi][ni]);
        }
      }
    }
    __syncthreads();
  }

#pragma unroll
  for (int mi = 0; mi < 2; ++mi) {
#pragma unroll
    for (int ni = 0; ni < 2; ++ni) {
#pragma unroll
      for (int r = 0; r < 8; ++r) {
        const int m = m0 + wm * 32 + mi * 16 + r + halfsel * 8;
        const int n = n0 + wn * 32 + ni * 16 + arow;
        const int b = m >> 11, t = m & (TSEQ - 1);
        const int h = n >> 6,  dd = n & (HDIM - 1);
        const _Float16 val = (_Float16)acc[mi][ni][r];
        const size_t bh = (size_t)b * NHEAD + h;
        if (which == 0)      Qh[(bh * TSEQ + t) * HDIM + dd] = val;
        else if (which == 1) Kh[(bh * TSEQ + t) * HDIM + dd] = val;
        else                 Vt[(bh * HDIM + dd) * TSEQ + t] = val;
      }
    }
  }
}

// ---------------------------------------------------------------------------
// Kernel 2: flash-style causal attention.
// grid = (B*H, T/128), block = 256 (8 waves); each wave owns 16 query rows.
// ---------------------------------------------------------------------------
__global__ __launch_bounds__(256) void attn_kernel(
    const _Float16* __restrict__ Qh, const _Float16* __restrict__ Kh,
    const _Float16* __restrict__ Vt, _Float16* __restrict__ Ao)
{
  __shared__ __align__(16) _Float16 Ps[8][16][32];

  const int bh   = blockIdx.x;
  const int tid  = threadIdx.x;
  const int wave = tid >> 5;
  const int lane = tid & 31;
  const int qb   = blockIdx.y * 128 + wave * 16;
  const int b    = bh >> 4, h = bh & 15;

  const _Float16* Qp = Qh + (size_t)bh * TSEQ * HDIM;
  const _Float16* Kp = Kh + (size_t)bh * TSEQ * HDIM;
  const _Float16* Vp = Vt + (size_t)bh * HDIM * TSEQ;

  const int arow    = lane & 15;
  const int halfsel = lane >> 4;
  const int akb     = halfsel ? 8 : 0;
  const int bkb     = halfsel ? 8 : 0;

  const _Float16* qrow = Qp + (size_t)(qb + arow) * HDIM;
  const v16h qf0 = ld_frag(qrow + akb,      qrow + akb + 16);
  const v16h qf1 = ld_frag(qrow + 32 + akb, qrow + 32 + akb + 16);

  v8f o[4] = {};
  float mrow[8], lrow[8];
#pragma unroll
  for (int r = 0; r < 8; ++r) { mrow[r] = -1e30f; lrow[r] = 0.0f; }

  const float scale = 0.125f;
  const int nsteps = (qb + 15) / 32 + 1;

  for (int j = 0; j < nsteps; ++j) {
    const int kk0 = j * 32;

    v8f s[2];
#pragma unroll
    for (int ni = 0; ni < 2; ++ni) {
      const _Float16* krow = Kp + (size_t)(kk0 + ni * 16 + arow) * HDIM;
      v16h kb0 = ld_frag(krow + bkb,      krow + bkb + 16);
      v16h kb1 = ld_frag(krow + 32 + bkb, krow + 32 + bkb + 16);
      v8f sv = {};
      sv = wmma16(qf0, kb0, sv);
      sv = wmma16(qf1, kb1, sv);
      s[ni] = sv;
    }

#pragma unroll
    for (int ni = 0; ni < 2; ++ni) {
#pragma unroll
      for (int r = 0; r < 8; ++r) {
        const int qg = qb + r + halfsel * 8;
        const int kg = kk0 + ni * 16 + arow;
        const float v = s[ni][r] * scale;
        s[ni][r] = (kg <= qg) ? v : -1e30f;
      }
    }

#pragma unroll
    for (int r = 0; r < 8; ++r) {
      float v = fmaxf(s[0][r], s[1][r]);
      v = fmaxf(v, __shfl_xor(v, 1, 16));
      v = fmaxf(v, __shfl_xor(v, 2, 16));
      v = fmaxf(v, __shfl_xor(v, 4, 16));
      v = fmaxf(v, __shfl_xor(v, 8, 16));
      const float mnew = fmaxf(mrow[r], v);
      const float corr = __expf(mrow[r] - mnew);
      mrow[r] = mnew;

      float rsum = 0.0f;
#pragma unroll
      for (int ni = 0; ni < 2; ++ni) {
        const float p = __expf(s[ni][r] - mnew);
        rsum += p;
        Ps[wave][r + halfsel * 8][ni * 16 + arow] = (_Float16)p;
      }
      rsum += __shfl_xor(rsum, 1, 16);
      rsum += __shfl_xor(rsum, 2, 16);
      rsum += __shfl_xor(rsum, 4, 16);
      rsum += __shfl_xor(rsum, 8, 16);
      lrow[r] = lrow[r] * corr + rsum;
#pragma unroll
      for (int nt = 0; nt < 4; ++nt) o[nt][r] *= corr;
    }

    const _Float16* prow = &Ps[wave][arow][0];
    const v16h pf = ld_frag(prow + akb, prow + akb + 16);
#pragma unroll
    for (int nt = 0; nt < 4; ++nt) {
      const _Float16* vrow = Vp + (size_t)(nt * 16 + arow) * TSEQ + kk0;
      v16h vf = ld_frag(vrow + bkb, vrow + bkb + 16);
      o[nt] = wmma16(pf, vf, o[nt]);
    }
  }

#pragma unroll
  for (int nt = 0; nt < 4; ++nt) {
#pragma unroll
    for (int r = 0; r < 8; ++r) {
      const int q  = qb + r + halfsel * 8;
      const int dd = nt * 16 + arow;
      const float val = o[nt][r] / lrow[r];
      Ao[((size_t)(b * TSEQ + q)) * CDIM + h * HDIM + dd] = (_Float16)val;
    }
  }
}

// ---------------------------------------------------------------------------
// Kernel 3: output projection, out = A Wo^T + bo (f32 out), TDM-staged.
// ---------------------------------------------------------------------------
__global__ __launch_bounds__(256) void out_proj_kernel(
    const _Float16* __restrict__ Ao, const _Float16* __restrict__ Woh,
    const float* __restrict__ bo, float* __restrict__ out)
{
  __shared__ __align__(16) _Float16 As[2][128][KSTAGE];
  __shared__ __align__(16) _Float16 Bs[2][64][KSTAGE];

  const int n0 = blockIdx.x * 64;
  const int m0 = blockIdx.y * 128;
  const int tid  = threadIdx.x;
  const int wave = tid >> 5;
  const int lane = tid & 31;
  const int wm = wave & 3;
  const int wn = wave >> 2;

  const int arow    = lane & 15;
  const int halfsel = lane >> 4;
  const int akb     = halfsel ? 8 : 0;
  const int bkb     = halfsel ? 8 : 0;

  if (wave == 0) {
    tdm_load_2d((unsigned)(size_t)&As[0][0][0], Ao + (size_t)m0 * CDIM,
                CDIM, 128, CDIM, KSTAGE, 128);
    tdm_load_2d((unsigned)(size_t)&Bs[0][0][0], Woh + (size_t)n0 * CDIM,
                CDIM, 64, CDIM, KSTAGE, 64);
  }

  v8f acc[2][2] = {};

  for (int kt = 0; kt < CDIM / KSTAGE; ++kt) {
    const int buf = kt & 1;
    if (wave == 0) {
      if (kt + 1 < CDIM / KSTAGE) {
        const int k1 = (kt + 1) * KSTAGE;
        tdm_load_2d((unsigned)(size_t)&As[buf ^ 1][0][0],
                    Ao + (size_t)m0 * CDIM + k1, CDIM, 128, CDIM, KSTAGE, 128);
        tdm_load_2d((unsigned)(size_t)&Bs[buf ^ 1][0][0],
                    Woh + (size_t)n0 * CDIM + k1, CDIM, 64, CDIM, KSTAGE, 64);
        __builtin_amdgcn_s_wait_tensorcnt(2);
      } else {
        __builtin_amdgcn_s_wait_tensorcnt(0);
      }
    }
    __syncthreads();

#pragma unroll
    for (int ks = 0; ks < 2; ++ks) {
#pragma unroll
      for (int mi = 0; mi < 2; ++mi) {
        const _Float16* ap = &As[buf][wm * 32 + mi * 16 + arow][ks * 32];
        v16h a = ld_frag(ap + akb, ap + akb + 16);
#pragma unroll
        for (int ni = 0; ni < 2; ++ni) {
          const _Float16* bp = &Bs[buf][wn * 32 + ni * 16 + arow][ks * 32];
          v16h b = ld_frag(bp + bkb, bp + bkb + 16);
          acc[mi][ni] = wmma16(a, b, acc[mi][ni]);
        }
      }
    }
    __syncthreads();
  }

#pragma unroll
  for (int mi = 0; mi < 2; ++mi) {
#pragma unroll
    for (int ni = 0; ni < 2; ++ni) {
#pragma unroll
      for (int r = 0; r < 8; ++r) {
        const int m = m0 + wm * 32 + mi * 16 + r + halfsel * 8;
        const int n = n0 + wn * 32 + ni * 16 + arow;
        out[(size_t)m * CDIM + n] = acc[mi][ni][r] + bo[n];
      }
    }
  }
}

// ---------------------------------------------------------------------------
extern "C" void kernel_launch(void* const* d_in, const int* in_sizes, int n_in,
                              void* d_out, int out_size, void* d_ws, size_t ws_size,
                              hipStream_t stream) {
  const float* x  = (const float*)d_in[0];
  const float* Wq = (const float*)d_in[1];
  const float* Wk = (const float*)d_in[2];
  const float* Wv = (const float*)d_in[3];
  const float* Wo = (const float*)d_in[4];
  const float* bo = (const float*)d_in[5];
  float* out = (float*)d_out;

  const size_t tokelems = (size_t)NTOK * CDIM;   // 4096*1024
  const size_t welems   = (size_t)CDIM * CDIM;   // 1024*1024
  _Float16* xh  = (_Float16*)d_ws;
  _Float16* Wqh = xh  + tokelems;
  _Float16* Wkh = Wqh + welems;
  _Float16* Wvh = Wkh + welems;
  _Float16* Woh = Wvh + welems;
  _Float16* Qh  = Woh + welems;
  _Float16* Kh  = Qh  + tokelems;
  _Float16* Vt  = Kh  + tokelems;
  _Float16* Ao  = Vt  + tokelems;

  convert_f32_f16_kernel<<<dim3(2048 + 4 * 512), 256, 0, stream>>>(
      x, Wq, Wk, Wv, Wo, xh, Wqh, Wkh, Wvh, Woh);
  qkv_gemm_kernel<<<dim3(CDIM / 64, NTOK / 128, 3), 256, 0, stream>>>(
      xh, Wqh, Wkh, Wvh, Qh, Kh, Vt);
  attn_kernel<<<dim3(2 * NHEAD, TSEQ / 128), 256, 0, stream>>>(Qh, Kh, Vt, Ao);
  out_proj_kernel<<<dim3(CDIM / 64, NTOK / 128), 256, 0, stream>>>(Ao, Woh, bo, out);
}
